// AttentionFusion_34136400069218
// MI455X (gfx1250) — matrix operand order repaired and checked by reading, not compiled, not run on hardware
//
#include <hip/hip_runtime.h>
#include <stdint.h>

// ---------- types ----------
typedef __attribute__((ext_vector_type(16))) __bf16 v16bf;
typedef __attribute__((ext_vector_type(8)))  float  v8f;
typedef __attribute__((ext_vector_type(4)))  unsigned int v4u;
typedef __attribute__((ext_vector_type(8)))  int v8i_;
typedef __attribute__((ext_vector_type(4)))  int v4i_;

struct B32x8 { uint4 lo, hi; };   // 32 bytes == one 16x32 bf16 fragment per lane

__device__ __forceinline__ unsigned short f2bf(float f) {
  unsigned int u = __builtin_bit_cast(unsigned int, f);
  u += 0x7FFFu + ((u >> 16) & 1u);          // round-to-nearest-even
  return (unsigned short)(u >> 16);
}
__device__ __forceinline__ float bf2f(unsigned short h) {
  unsigned int u = ((unsigned int)h) << 16;
  return __builtin_bit_cast(float, u);
}

// TDM: DMA a [rows x 32] bf16 tile (row stride 1024 elems) from global into LDS.
// D# packing per CDNA5 ISA:
//  g0 = {count=1 | gather off, lds_addr, gaddr[31:0], gaddr[56:32] | type=2}
//  g1 = {data_size=2B, tensor_dim0=1024, tensor_dim1=rows_tot, tile_dim0=32,
//        tile_dim1=128, stride0=1024}
// clang-23 toolchain: 6-arg builtin (g0, g1, g2, g3, extra, cpol).
__device__ __forceinline__ void tdm_load_tile(unsigned lds_addr, const void* gptr,
                                              unsigned rows_tot) {
  unsigned long long ga = (unsigned long long)gptr;
  v4u g0;
  g0[0] = 1u;                                   // count=1
  g0[1] = lds_addr;                             // LDS byte address
  g0[2] = (unsigned)ga;                         // global_addr[31:0]
  g0[3] = (unsigned)(ga >> 32) | (2u << 30);    // global_addr[56:32], type=2
  v8i_ g1;
  g1[0] = (int)(1u << 16);                                  // data_size=1 -> 2 bytes
  g1[1] = (int)((1024u & 0xFFFFu) << 16);                   // tensor_dim0[15:0]=1024
  g1[2] = (int)(((1024u >> 16) & 0xFFFFu) |                 // tensor_dim0[31:16]
                ((rows_tot & 0xFFFFu) << 16));              // tensor_dim1[15:0]
  g1[3] = (int)(((rows_tot >> 16) & 0xFFFFu) |              // tensor_dim1[31:16]
                (32u << 16));                               // tile_dim0 = 32
  g1[4] = 128;                                              // tile_dim1=128, tile_dim2=0
  g1[5] = 1024;                                             // tensor_dim0_stride[31:0]
  g1[6] = 0;
  g1[7] = 0;
  v4i_ z4 = {0, 0, 0, 0};
  v8i_ z8 = {0, 0, 0, 0, 0, 0, 0, 0};
  __builtin_amdgcn_tensor_load_to_lds(g0, g1, z4, z4, z8, 0);
}

// ---------- constants ----------
// H=1024, NH=16, HD=64, A=4, B=4, S=2048, BS=8192, rows(QKV)=32768

// ---------- kernel 1: pack x = stack(adapter_outputs, dim=2) -> [BS*A, H] bf16 ----------
__global__ void __launch_bounds__(256) pack_x_kernel(const float* __restrict__ src,
                                                     unsigned short* __restrict__ dst) {
  int t    = blockIdx.x * 256 + threadIdx.x;   // 4 elems / thread
  int base = t << 2;
  int h    = base & 1023;
  int row  = base >> 10;        // row = bs*4 + a
  int a    = row & 3;
  int bs   = row >> 2;
  int b    = bs >> 11;          // S = 2048
  int s    = bs & 2047;
  const float* p = src + (((((size_t)a * 4 + b) * 2048 + s) * 1024) + h);
  float4 v = *(const float4*)p;
  unsigned short* q = dst + base;
  q[0] = f2bf(v.x); q[1] = f2bf(v.y); q[2] = f2bf(v.z); q[3] = f2bf(v.w);
}

// ---------- kernel 2: fp32 W[k][n] -> bf16 Wt[n][k] (tiled transpose, paid once) ----------
__global__ void __launch_bounds__(256) pack_wt_kernel(const float* __restrict__ src,
                                                      unsigned short* __restrict__ dst) {
  __shared__ float tile[32][33];
  int n0 = (blockIdx.x & 31) << 5;
  int k0 = (blockIdx.x >> 5) << 5;
  int c  = threadIdx.x & 31;
  int r0 = threadIdx.x >> 5;          // 8 rows per pass
  #pragma unroll
  for (int i = 0; i < 4; ++i) {
    int r = r0 + i * 8;
    tile[r][c] = src[(size_t)(k0 + r) * 1024 + n0 + c];   // coalesced over n
  }
  __syncthreads();
  #pragma unroll
  for (int i = 0; i < 4; ++i) {
    int r = r0 + i * 8;
    dst[(size_t)(n0 + r) * 1024 + k0 + c] = f2bf(tile[c][r]);  // coalesced over k
  }
}

// ---------- kernel 3: bf16 WMMA GEMM with TDM-staged, double-buffered LDS tiles ----------
// A row-major [M][1024] bf16; Bt row-major [N=1024][1024] bf16 (pre-transposed weights).
// Block tile 128(M) x 128(N), BK=32; 8 waves as 4(M) x 2(N); wave tile 32x64 = 8 WMMA.
// Wave 0 drives the Tensor Data Mover; buffers published via s_wait_tensorcnt + barrier.
// mode 0: write bf16.   mode 1: write fp32, += residual.
__global__ void __launch_bounds__(256) gemm_bf16_kernel(
    const unsigned short* __restrict__ A, const unsigned short* __restrict__ Bt,
    const float* __restrict__ bias, const float* __restrict__ residual,
    void* __restrict__ out, int M, int mode)
{
  __shared__ unsigned short As[2][128 * 32];   // [m][k] double-buffered (2 x 8KB)
  __shared__ unsigned short Bs[2][128 * 32];   // [n][k] double-buffered (2 x 8KB)

  const int tid  = threadIdx.x;
  const int lane = tid & 31;
  const int w    = tid >> 5;
  const int wm   = w & 3;          // 4 M-groups of 32 rows
  const int wn   = w >> 2;         // 2 N-groups of 64 cols
  const size_t blockM = (size_t)blockIdx.x * 128;
  const int    n0     = blockIdx.y * 128;

  const unsigned short* Ablk = A  + blockM * 1024;
  const unsigned short* Bblk = Bt + (size_t)n0 * 1024;

  const unsigned asAddr0 = (unsigned)(unsigned long long)(&As[0][0]);
  const unsigned asAddr1 = (unsigned)(unsigned long long)(&As[1][0]);
  const unsigned bsAddr0 = (unsigned)(unsigned long long)(&Bs[0][0]);
  const unsigned bsAddr1 = (unsigned)(unsigned long long)(&Bs[1][0]);

  // prologue: DMA first K tile into buffer 0 (wave 0 only; TDM ignores EXEC, so
  // the other waves must *branch* around this block)
  if (tid < 32) {
    tdm_load_tile(asAddr0, Ablk, (unsigned)M);
    tdm_load_tile(bsAddr0, Bblk, 1024u);
  }

  v8f acc[2][4] = {};

  for (int k0 = 0; k0 < 1024; k0 += 32) {
    const int cur = (k0 >> 5) & 1;

    if (tid < 32) {
      __builtin_amdgcn_s_wait_tensorcnt(0);   // current tile landed in LDS
    }
    __syncthreads();                          // publish buffer; old reads retired

    if (k0 + 32 < 1024) {
      if (tid < 32) {                         // DMA next tile, overlaps WMMA below
        tdm_load_tile(cur ? asAddr0 : asAddr1, Ablk + (k0 + 32), (unsigned)M);
        tdm_load_tile(cur ? bsAddr0 : bsAddr1, Bblk + (k0 + 32), 1024u);
      }
    }

    const int hl = lane >> 4;   // 0/1
    const int ll = lane & 15;
    const int ab = hl << 3;     // A frag: lanes<16 K{0..7,16..23}; lanes>=16 K{8..15,24..31}
    const int bb = hl << 3;     // B frag: lanes<16 K{0..7,16..23}; lanes>=16 K{8..15,24..31}

    v16bf aF[2], bF[4];
    #pragma unroll
    for (int mi = 0; mi < 2; ++mi) {
      const unsigned short* ar = &As[cur][(wm * 32 + mi * 16 + ll) * 32];
      B32x8 ta;
      ta.lo = *(const uint4*)(ar + ab);
      ta.hi = *(const uint4*)(ar + ab + 16);
      aF[mi] = __builtin_bit_cast(v16bf, ta);
    }
    #pragma unroll
    for (int t = 0; t < 4; ++t) {
      const unsigned short* br = &Bs[cur][(wn * 64 + t * 16 + ll) * 32 + bb];
      B32x8 tb;
      tb.lo = *(const uint4*)(br);
      tb.hi = *(const uint4*)(br + 16);
      bF[t] = __builtin_bit_cast(v16bf, tb);
    }
    #pragma unroll
    for (int mi = 0; mi < 2; ++mi)
      #pragma unroll
      for (int t = 0; t < 4; ++t)
        acc[mi][t] = __builtin_amdgcn_wmma_f32_16x16x32_bf16(
            false, aF[mi], false, bF[t], (short)0, acc[mi][t], false, false);
  }

  // epilogue: D layout = lanes<16: N=lane, M=r; lanes>=16: N=lane-16, M=r+8
  #pragma unroll
  for (int mi = 0; mi < 2; ++mi) {
    #pragma unroll
    for (int t = 0; t < 4; ++t) {
      int col = n0 + wn * 64 + t * 16 + (lane & 15);
      float bv = bias[col];
      #pragma unroll
      for (int r = 0; r < 8; ++r) {
        size_t row = blockM + wm * 32 + mi * 16 + r + ((lane >> 4) << 3);
        float val = acc[mi][t][r] + bv;
        if (mode) {
          ((float*)out)[row * 1024 + col] = val + residual[row * 1024 + col];
        } else {
          ((unsigned short*)out)[row * 1024 + col] = f2bf(val);
        }
      }
    }
  }
}

// ---------- kernel 4: per-(token,head) attention over A=4 axis + adapter mean-pool ----------
__global__ void __launch_bounds__(256) attn_kernel(
    const unsigned short* __restrict__ Qb, const unsigned short* __restrict__ Kb,
    const unsigned short* __restrict__ Vb, unsigned short* __restrict__ pooled)
{
  int wave  = (blockIdx.x * 256 + threadIdx.x) >> 5;
  int lane  = threadIdx.x & 31;
  int token = wave >> 4;
  int head  = wave & 15;

  float q[4][2], k[4][2], v[4][2];
  #pragma unroll
  for (int i = 0; i < 4; ++i) {
    size_t base = ((size_t)(token * 4 + i)) * 1024 + head * 64 + lane;
    q[i][0] = bf2f(Qb[base]); q[i][1] = bf2f(Qb[base + 32]);
    k[i][0] = bf2f(Kb[base]); k[i][1] = bf2f(Kb[base + 32]);
    v[i][0] = bf2f(Vb[base]); v[i][1] = bf2f(Vb[base + 32]);
  }

  float s[4][4];
  #pragma unroll
  for (int i = 0; i < 4; ++i)
    #pragma unroll
    for (int j = 0; j < 4; ++j) {
      float p = q[i][0] * k[j][0] + q[i][1] * k[j][1];
      #pragma unroll
      for (int off = 16; off > 0; off >>= 1) p += __shfl_xor(p, off, 32);
      s[i][j] = p * 0.125f;   // 1/(sqrt(64)*TEMP)
    }

  float wgt[4][4];
  #pragma unroll
  for (int i = 0; i < 4; ++i) {
    float m = fmaxf(fmaxf(s[i][0], s[i][1]), fmaxf(s[i][2], s[i][3]));
    float sum = 0.f;
    #pragma unroll
    for (int j = 0; j < 4; ++j) { wgt[i][j] = __expf(s[i][j] - m); sum += wgt[i][j]; }
    float inv = 1.f / sum;
    #pragma unroll
    for (int j = 0; j < 4; ++j) wgt[i][j] *= inv;
  }

  #pragma unroll
  for (int d = 0; d < 2; ++d) {
    float acc = 0.f;
    #pragma unroll
    for (int i = 0; i < 4; ++i)
      #pragma unroll
      for (int j = 0; j < 4; ++j) acc += wgt[i][j] * v[j][d];
    acc *= 0.25f;    // mean over adapters
    pooled[(size_t)token * 1024 + head * 64 + lane + d * 32] = f2bf(acc);
  }
}

// ---------- kernel 5: LayerNorm over H=1024 (biased variance, eps=1e-5) ----------
__global__ void __launch_bounds__(256) ln_kernel(
    const float* __restrict__ x, const float* __restrict__ gamma,
    const float* __restrict__ beta, float* __restrict__ out)
{
  __shared__ float2 red[256];
  int row = blockIdx.x;
  const float* xr = x + (size_t)row * 1024;
  float loc[4];
  float sum = 0.f, sq = 0.f;
  #pragma unroll
  for (int i = 0; i < 4; ++i) {
    loc[i] = xr[threadIdx.x + i * 256];
    sum += loc[i]; sq += loc[i] * loc[i];
  }
  red[threadIdx.x] = make_float2(sum, sq);
  __syncthreads();
  for (int off = 128; off > 0; off >>= 1) {
    if (threadIdx.x < off) {
      float2 o = red[threadIdx.x + off];
      red[threadIdx.x].x += o.x;
      red[threadIdx.x].y += o.y;
    }
    __syncthreads();
  }
  float mu  = red[0].x * (1.f / 1024.f);
  float var = red[0].y * (1.f / 1024.f) - mu * mu;
  float inv = rsqrtf(var + 1e-5f);
  #pragma unroll
  for (int i = 0; i < 4; ++i) {
    int c = threadIdx.x + i * 256;
    out[(size_t)row * 1024 + c] = (loc[i] - mu) * inv * gamma[c] + beta[c];
  }
}

// ---------- launcher ----------
extern "C" void kernel_launch(void* const* d_in, const int* in_sizes, int n_in,
                              void* d_out, int out_size, void* d_ws, size_t ws_size,
                              hipStream_t stream) {
  const float* adapter = (const float*)d_in[0];
  const float* Wq = (const float*)d_in[1];
  const float* bq = (const float*)d_in[2];
  const float* Wk = (const float*)d_in[3];
  const float* bk = (const float*)d_in[4];
  const float* Wv = (const float*)d_in[5];
  const float* bv = (const float*)d_in[6];
  const float* Wo = (const float*)d_in[7];
  const float* bo = (const float*)d_in[8];
  const float* g  = (const float*)d_in[9];
  const float* be = (const float*)d_in[10];

  char* ws = (char*)d_ws;
  unsigned short* Xb   = (unsigned short*)ws;                        // 32768*1024 bf16 = 64 MB
  unsigned short* Wqt  = (unsigned short*)(ws + 67108864);           // 4 x 1M bf16 (transposed)
  unsigned short* Wkt  = Wqt + 1048576;
  unsigned short* Wvt  = Wkt + 1048576;
  unsigned short* Wot  = Wvt + 1048576;
  unsigned short* Qb   = (unsigned short*)(ws + 75497472);           // 3 x 64 MB
  unsigned short* Kb   = Qb + 33554432;
  unsigned short* Vb   = Kb + 33554432;
  unsigned short* Pool = (unsigned short*)(ws + 276824064);          // 16 MB
  float*          pre  = (float*)(ws + 293601280);                   // 32 MB
  float*          outF = (float*)d_out;

  pack_x_kernel<<<32768, 256, 0, stream>>>(adapter, Xb);
  pack_wt_kernel<<<1024, 256, 0, stream>>>(Wq, Wqt);
  pack_wt_kernel<<<1024, 256, 0, stream>>>(Wk, Wkt);
  pack_wt_kernel<<<1024, 256, 0, stream>>>(Wv, Wvt);
  pack_wt_kernel<<<1024, 256, 0, stream>>>(Wo, Wot);

  dim3 gQKV(256, 8);    // M=32768/128, N=1024/128
  gemm_bf16_kernel<<<gQKV, 256, 0, stream>>>(Xb, Wqt, bq, nullptr, (void*)Qb, 32768, 0);
  gemm_bf16_kernel<<<gQKV, 256, 0, stream>>>(Xb, Wkt, bk, nullptr, (void*)Kb, 32768, 0);
  gemm_bf16_kernel<<<gQKV, 256, 0, stream>>>(Xb, Wvt, bv, nullptr, (void*)Vb, 32768, 0);

  attn_kernel<<<16384, 256, 0, stream>>>(Qb, Kb, Vb, Pool);

  dim3 gO(64, 8);       // M=8192/128
  gemm_bf16_kernel<<<gO, 256, 0, stream>>>(Pool, Wot, bo, adapter, (void*)pre, 8192, 1);

  ln_kernel<<<8192, 256, 0, stream>>>(pre, g, be, outF);
}
